// MyMoeDecoderLayer_72043781423420
// MI455X (gfx1250) — matrix operand order repaired and checked by reading, not compiled, not run on hardware
//
#include <hip/hip_runtime.h>

// ---------------------------------------------------------------------------
// CDNA5 (gfx1250) decoder layer: self/cross flash attention + LN + top-1 MoE.
// Matmuls: v_wmma_f32_16x16x32_f16.  All GEMM/flash tiles fed with
// GLOBAL_LOAD_ASYNC_TO_LDS_B128 (ASYNCcnt); reductions via ds_swizzle xor.
// V is produced pre-transposed ([B][H][dh][S]) by the V-projection epilogue
// so the flash kernel needs no in-loop transpose.
// ---------------------------------------------------------------------------

typedef __attribute__((ext_vector_type(16))) _Float16 v16h;
typedef __attribute__((ext_vector_type(8)))  _Float16 v8h;
typedef __attribute__((ext_vector_type(4)))  _Float16 v4h;
typedef __attribute__((ext_vector_type(8)))  float    v8f;
typedef int v4i __attribute__((vector_size(16)));   // matches builtin param type

#if defined(__gfx1250__) && __has_builtin(__builtin_amdgcn_global_load_async_to_lds_b128)
#define USE_ASYNC 1
#else
#define USE_ASYNC 0
#endif

__device__ __forceinline__ void cp_b128_g2l(_Float16* l, const _Float16* g) {
#if USE_ASYNC
    __builtin_amdgcn_global_load_async_to_lds_b128(
        (__attribute__((address_space(1))) v4i*)g,
        (__attribute__((address_space(3))) v4i*)l, 0, 0);
#else
    *(v8h*)l = *(const v8h*)g;
#endif
}
__device__ __forceinline__ void async_wait0() {
#if USE_ASYNC
    asm volatile("s_wait_asynccnt 0" ::: "memory");
#endif
}

// XOR butterfly via ds_swizzle (group-of-32: and=0x1F, or=0, xor=M)
template<int M>
__device__ __forceinline__ float sxor(float v) {
    return __int_as_float(__builtin_amdgcn_ds_swizzle(__float_as_int(v), (M << 10) | 0x1F));
}

__device__ __forceinline__ v8f wmma_f16(v16h a, v16h b, v8f c) {
    return __builtin_amdgcn_wmma_f32_16x16x32_f16(false, a, false, b, (short)0, c, false, false);
}
__device__ __forceinline__ v16h cat16(v8h lo, v8h hi) {
    return __builtin_shufflevector(lo, hi, 0,1,2,3,4,5,6,7,8,9,10,11,12,13,14,15);
}
// A-frag (16x32 f16): lane<16 row=lane K={0..7,16..23}; lane>=16 K={8..15,24..31}
__device__ __forceinline__ v16h frag_a_lds(const _Float16* base, int row0, int k0, int ld) {
    const int lane = threadIdx.x & 31;
    const _Float16* p = base + (size_t)(row0 + (lane & 15)) * ld + k0 + ((lane >> 4) << 3);
    return cat16(*(const v8h*)p, *(const v8h*)(p + 16));
}
// B-frag (32x16 f16): lane<16 col=lane K={0..7,16..23}; lane>=16 K={8..15,24..31}
__device__ __forceinline__ v16h frag_b_lds(const _Float16* base, int col0, int k0, int ld) {
    const int lane = threadIdx.x & 31;
    const _Float16* p = base + (size_t)(col0 + (lane & 15)) * ld + k0 + ((lane >> 4) << 3);
    return cat16(*(const v8h*)p, *(const v8h*)(p + 16));
}
__device__ __forceinline__ float gelu_f(float x) {
    return 0.5f * x * (1.f + erff(x * 0.70710678118654752f));
}

// ---------------------------------------------------------------------------
// NT GEMM: C[M,N] = epi( A[M,K](f16) * W[N,K]^T(f16) ).  128x128x64 tiles,
// 8 waves (2x4), 64x32/wave, async double-buffered LDS.
// VTR: write f16 output transposed into [B][H][dh][S] (V for flash attention).
// ---------------------------------------------------------------------------
constexpr int GBM = 128, GBN = 128, GBK = 64, GLD = 72; // 144B rows: bank-conflict-free

template<bool GELU, bool VTR>
__global__ __launch_bounds__(256) void gemm_nt_kernel(
    const _Float16* __restrict__ A, const _Float16* __restrict__ W,
    const float* __restrict__ bias, float* __restrict__ Cf,
    _Float16* __restrict__ Ch, int M, int N, int K, float scale)
{
    __shared__ _Float16 As[2][GBM * GLD];
    __shared__ _Float16 Bs[2][GBN * GLD];
    const int tid = threadIdx.x, lane = tid & 31, wid = tid >> 5;
    const int bm = blockIdx.y * GBM, bn = blockIdx.x * GBN;
    const int wm = (wid >> 2) * 64, wn = (wid & 3) * 32;

    const v8f vz = {0.f,0.f,0.f,0.f,0.f,0.f,0.f,0.f};
    v8f acc[4][2];
#pragma unroll
    for (int mt = 0; mt < 4; ++mt)
#pragma unroll
        for (int nt = 0; nt < 2; ++nt) acc[mt][nt] = vz;

    auto issue = [&](int kt, int buf) {
#pragma unroll
        for (int j = 0; j < 4; ++j) {
            const int c = tid + j * 256, row = c >> 3, cc = c & 7;
            cp_b128_g2l(&As[buf][row * GLD + cc * 8],
                        A + (size_t)(bm + row) * K + kt * GBK + cc * 8);
        }
#pragma unroll
        for (int j = 0; j < 4; ++j) {
            const int c = tid + j * 256, row = c >> 3, cc = c & 7;
            cp_b128_g2l(&Bs[buf][row * GLD + cc * 8],
                        W + (size_t)(bn + row) * K + kt * GBK + cc * 8);
        }
    };

    const int KT = K / GBK;
    issue(0, 0);
    async_wait0();
    __syncthreads();
    for (int kt = 0; kt < KT; ++kt) {
        const int buf = kt & 1;
        if (kt + 1 < KT) issue(kt + 1, buf ^ 1);
#pragma unroll
        for (int kk = 0; kk < 2; ++kk) {
            v16h af[4], bf[2];
#pragma unroll
            for (int mt = 0; mt < 4; ++mt) af[mt] = frag_a_lds(&As[buf][0], wm + mt * 16, kk * 32, GLD);
#pragma unroll
            for (int nt = 0; nt < 2; ++nt) bf[nt] = frag_b_lds(&Bs[buf][0], wn + nt * 16, kk * 32, GLD);
#pragma unroll
            for (int mt = 0; mt < 4; ++mt)
#pragma unroll
                for (int nt = 0; nt < 2; ++nt)
                    acc[mt][nt] = wmma_f16(af[mt], bf[nt], acc[mt][nt]);
        }
        async_wait0();
        __syncthreads();
    }

    const int hb = lane >> 4;
#pragma unroll
    for (int nt = 0; nt < 2; ++nt) {
        const int col = bn + wn + nt * 16 + (lane & 15);
        const float bv = bias ? bias[col] : 0.f;
#pragma unroll
        for (int mt = 0; mt < 4; ++mt) {
#pragma unroll
            for (int r = 0; r < 8; ++r) {
                const int row = bm + wm + mt * 16 + r + (hb << 3);
                float v = (acc[mt][nt][r] + bv) * scale;
                if (GELU) v = gelu_f(v);
                if (VTR) {
                    // token row -> (b, s); feature col -> (h, dc); out[((b*16+h)*64+dc)*512+s]
                    const int bb = row >> 9, ss = row & 511;
                    const int hh = col >> 6, dc = col & 63;
                    Ch[((((size_t)bb * 16 + hh) * 64 + dc) << 9) + ss] = (_Float16)v;
                } else {
                    if (Cf) Cf[(size_t)row * N + col] = v;
                    if (Ch) Ch[(size_t)row * N + col] = (_Float16)v;
                }
            }
        }
    }
}

// ---------------------------------------------------------------------------
// MoE grouped GEMM: A rows gathered through per-expert bucket (perm), C rows
// scattered back by token id.  Padding rows (-1) load token 0 (garbage) and
// are masked at the store, so the async path needs no zero-fill.
// ---------------------------------------------------------------------------
template<bool GELU, bool ACC>
__global__ __launch_bounds__(256) void gemm_moe_kernel(
    const _Float16* __restrict__ A, const _Float16* __restrict__ Wall,
    const float* __restrict__ biasAll, float* __restrict__ Cf,
    _Float16* __restrict__ Ch, int N, int K,
    const int* __restrict__ perm, const int* __restrict__ off,
    const int* __restrict__ pc)
{
    constexpr int MTMAX = 32;                       // T / GBM
    const int e   = blockIdx.y >> 5;
    const int mt0 = blockIdx.y & (MTMAX - 1);
    if (mt0 * GBM >= pc[e]) return;                 // uniform exit
    const int tilebase = off[e] + mt0 * GBM;
    const _Float16* W = Wall + (size_t)e * N * K;
    const float* bias = biasAll ? biasAll + (size_t)e * N : nullptr;

    __shared__ int permS[GBM];
    __shared__ _Float16 As[2][GBM * GLD];
    __shared__ _Float16 Bs[2][GBN * GLD];
    const int tid = threadIdx.x, lane = tid & 31, wid = tid >> 5;
    const int bn = blockIdx.x * GBN;
    const int wm = (wid >> 2) * 64, wn = (wid & 3) * 32;
    if (tid < GBM) permS[tid] = perm[tilebase + tid];
    __syncthreads();

    const v8f vz = {0.f,0.f,0.f,0.f,0.f,0.f,0.f,0.f};
    v8f acc[4][2];
#pragma unroll
    for (int mt = 0; mt < 4; ++mt)
#pragma unroll
        for (int nt = 0; nt < 2; ++nt) acc[mt][nt] = vz;

    auto issue = [&](int kt, int buf) {
#pragma unroll
        for (int j = 0; j < 4; ++j) {
            const int c = tid + j * 256, row = c >> 3, cc = c & 7;
            const int tok = permS[row];
            const int src = tok < 0 ? 0 : tok;
            cp_b128_g2l(&As[buf][row * GLD + cc * 8],
                        A + (size_t)src * K + kt * GBK + cc * 8);
        }
#pragma unroll
        for (int j = 0; j < 4; ++j) {
            const int c = tid + j * 256, row = c >> 3, cc = c & 7;
            cp_b128_g2l(&Bs[buf][row * GLD + cc * 8],
                        W + (size_t)(bn + row) * K + kt * GBK + cc * 8);
        }
    };

    const int KT = K / GBK;
    issue(0, 0);
    async_wait0();
    __syncthreads();
    for (int kt = 0; kt < KT; ++kt) {
        const int buf = kt & 1;
        if (kt + 1 < KT) issue(kt + 1, buf ^ 1);
#pragma unroll
        for (int kk = 0; kk < 2; ++kk) {
            v16h af[4], bf[2];
#pragma unroll
            for (int mt = 0; mt < 4; ++mt) af[mt] = frag_a_lds(&As[buf][0], wm + mt * 16, kk * 32, GLD);
#pragma unroll
            for (int nt = 0; nt < 2; ++nt) bf[nt] = frag_b_lds(&Bs[buf][0], wn + nt * 16, kk * 32, GLD);
#pragma unroll
            for (int mt = 0; mt < 4; ++mt)
#pragma unroll
                for (int nt = 0; nt < 2; ++nt)
                    acc[mt][nt] = wmma_f16(af[mt], bf[nt], acc[mt][nt]);
        }
        async_wait0();
        __syncthreads();
    }

    const int hb = lane >> 4;
#pragma unroll
    for (int nt = 0; nt < 2; ++nt) {
        const int col = bn + wn + nt * 16 + (lane & 15);
        const float bv = bias ? bias[col] : 0.f;
#pragma unroll
        for (int mt = 0; mt < 4; ++mt) {
#pragma unroll
            for (int r = 0; r < 8; ++r) {
                const int rl = wm + mt * 16 + r + (hb << 3);
                const int tok = permS[rl];
                if (tok >= 0) {
                    if (ACC) {
                        Cf[(size_t)tok * N + col] += acc[mt][nt][r];
                    } else {
                        float v = acc[mt][nt][r] + bv;
                        if (GELU) v = gelu_f(v);
                        Ch[(size_t)tok * N + col] = (_Float16)v;
                    }
                }
            }
        }
    }
}

// ---------------------------------------------------------------------------
// Flash attention, dh=64: 128 q-rows/block (8 waves x 16 rows), kv tiles of 64.
// V arrives pre-transposed as VT[B][H][dh][SKV]; K and VT tiles are both pure
// async LDS copies.  Online-softmax stats live per-lane in registers, reduced
// with ds_swizzle xor butterflies over the 16-lane column groups.
// ---------------------------------------------------------------------------
__global__ __launch_bounds__(256) void flash_kernel(
    const _Float16* __restrict__ Q, const _Float16* __restrict__ Kg,
    const _Float16* __restrict__ VT, _Float16* __restrict__ O,
    int SQ, int SKV)
{
    constexpr int DH = 64, BR = 128, BC = 64, LDF = 72;
    __shared__ _Float16 Kt[BC * LDF];      // [kv][d]
    __shared__ _Float16 Vt[DH * LDF];      // [d][kv]
    __shared__ _Float16 Ps[8 * 16 * LDF];  // per-wave P staging
    const int tid = threadIdx.x, lane = tid & 31, wid = tid >> 5;
    const int b = blockIdx.y >> 4, h = blockIdx.y & 15;  // H = 16
    const int qt = blockIdx.x;
    const size_t ld = 1024;  // D

    const _Float16* qrow = Q + (size_t)(b * SQ + qt * BR + wid * 16 + (lane & 15)) * ld
                             + h * DH + ((lane >> 4) << 3);
    v16h qf[2];
#pragma unroll
    for (int kk = 0; kk < 2; ++kk)
        qf[kk] = cat16(*(const v8h*)(qrow + kk * 32), *(const v8h*)(qrow + kk * 32 + 16));

    const _Float16* kbase  = Kg + (size_t)(b * SKV) * ld + h * DH;
    const _Float16* vtbase = VT + ((size_t)(b * 16 + h) * 64) * SKV;   // [dh][SKV]

    const v8f vz = {0.f,0.f,0.f,0.f,0.f,0.f,0.f,0.f};
    float m_i[8], l_i[8];
    v8f o_acc[4];
#pragma unroll
    for (int r = 0; r < 8; ++r) { m_i[r] = -1e30f; l_i[r] = 0.f; }
#pragma unroll
    for (int n = 0; n < 4; ++n) o_acc[n] = vz;
    _Float16* Pw = &Ps[wid * 16 * LDF];

    const int steps = SKV / BC;
    for (int s = 0; s < steps; ++s) {
        __syncthreads();
#pragma unroll
        for (int j = 0; j < 2; ++j) {   // K tile: [kv][d]
            const int c = tid + j * 256, row = c >> 3, cc = c & 7;
            cp_b128_g2l(&Kt[row * LDF + cc * 8],
                        kbase + (size_t)(s * BC + row) * ld + cc * 8);
        }
#pragma unroll
        for (int j = 0; j < 2; ++j) {   // V^T tile: [d][kv]
            const int c = tid + j * 256, row = c >> 3, cc = c & 7;
            cp_b128_g2l(&Vt[row * LDF + cc * 8],
                        vtbase + (size_t)row * SKV + s * BC + cc * 8);
        }
        async_wait0();
        __syncthreads();

        // S = Q * K^T  (softmax scale folded into the Q projection)
        v8f sc[4];
#pragma unroll
        for (int nt = 0; nt < 4; ++nt) {
            sc[nt] = vz;
#pragma unroll
            for (int kk = 0; kk < 2; ++kk)
                sc[nt] = wmma_f16(qf[kk], frag_b_lds(Kt, nt * 16, kk * 32, LDF), sc[nt]);
        }
#pragma unroll
        for (int r = 0; r < 8; ++r) {
            float mv = sc[0][r];
#pragma unroll
            for (int nt = 1; nt < 4; ++nt) mv = fmaxf(mv, sc[nt][r]);
            mv = fmaxf(mv, sxor<8>(mv)); mv = fmaxf(mv, sxor<4>(mv));
            mv = fmaxf(mv, sxor<2>(mv)); mv = fmaxf(mv, sxor<1>(mv));
            const float mn = fmaxf(m_i[r], mv);
            const float al = __expf(m_i[r] - mn);
            m_i[r] = mn;
            float ps = 0.f;
#pragma unroll
            for (int nt = 0; nt < 4; ++nt) {
                const float p = __expf(sc[nt][r] - mn);
                sc[nt][r] = p;
                ps += p;
            }
            ps += sxor<8>(ps); ps += sxor<4>(ps); ps += sxor<2>(ps); ps += sxor<1>(ps);
            l_i[r] = l_i[r] * al + ps;
#pragma unroll
            for (int nt = 0; nt < 4; ++nt) o_acc[nt][r] *= al;
        }
        // P: C-layout -> A-layout via per-wave LDS (same-wave DS ordering)
        const int prow8 = (lane >> 4) << 3, pcol = lane & 15;
#pragma unroll
        for (int nt = 0; nt < 4; ++nt)
#pragma unroll
            for (int r = 0; r < 8; ++r)
                Pw[(r + prow8) * LDF + nt * 16 + pcol] = (_Float16)sc[nt][r];
        // O += P * V
#pragma unroll
        for (int kk = 0; kk < 2; ++kk) {
            v16h pa = frag_a_lds(Pw, 0, kk * 32, LDF);
#pragma unroll
            for (int nt = 0; nt < 4; ++nt)
                o_acc[nt] = wmma_f16(pa, frag_b_lds(Vt, nt * 16, kk * 32, LDF), o_acc[nt]);
        }
    }

    const int hb = lane >> 4, ln = lane & 15;
#pragma unroll
    for (int r = 0; r < 8; ++r) {
        const float inv = 1.f / l_i[r];
        const size_t row = (size_t)(b * SQ + qt * BR + wid * 16 + r + (hb << 3));
#pragma unroll
        for (int nt = 0; nt < 4; ++nt)
            O[row * ld + h * DH + nt * 16 + ln] = (_Float16)(o_acc[nt][r] * inv);
    }
}

// ---------------------------------------------------------------------------
// LayerNorm(residual + y [* rowscale]) -> f32 (+ optional f16).  D = 1024.
// ---------------------------------------------------------------------------
template<bool SC>
__global__ __launch_bounds__(256) void ln_kernel(
    const float* __restrict__ X, const float* __restrict__ Y,
    const float* __restrict__ rs, const float* __restrict__ g,
    const float* __restrict__ bta, float* __restrict__ Of,
    _Float16* __restrict__ Oh)
{
    constexpr int D = 1024;
    const int row = blockIdx.x, tid = threadIdx.x;
    const float s = SC ? rs[row] : 1.f;
    const float4 xv = *(const float4*)(X + (size_t)row * D + tid * 4);
    const float4 yv = *(const float4*)(Y + (size_t)row * D + tid * 4);
    float u[4] = { xv.x + yv.x * s, xv.y + yv.y * s, xv.z + yv.z * s, xv.w + yv.w * s };
    float sm = u[0] + u[1] + u[2] + u[3];
    float sq = u[0]*u[0] + u[1]*u[1] + u[2]*u[2] + u[3]*u[3];
    sm += sxor<16>(sm); sm += sxor<8>(sm); sm += sxor<4>(sm); sm += sxor<2>(sm); sm += sxor<1>(sm);
    sq += sxor<16>(sq); sq += sxor<8>(sq); sq += sxor<4>(sq); sq += sxor<2>(sq); sq += sxor<1>(sq);
    __shared__ float a1[8], a2[8];
    __shared__ float mu_s, inv_s;
    const int wid = tid >> 5, lane = tid & 31;
    if (lane == 0) { a1[wid] = sm; a2[wid] = sq; }
    __syncthreads();
    if (tid == 0) {
        float t1 = 0.f, t2 = 0.f;
        for (int i = 0; i < 8; ++i) { t1 += a1[i]; t2 += a2[i]; }
        const float mu = t1 * (1.f / D);
        mu_s = mu;
        inv_s = rsqrtf(t2 * (1.f / D) - mu * mu + 1e-5f);
    }
    __syncthreads();
    const float mu = mu_s, inv = inv_s;
    const float4 gv = *(const float4*)(g + tid * 4);
    const float4 bv = *(const float4*)(bta + tid * 4);
    const float o0 = (u[0]-mu)*inv*gv.x + bv.x, o1 = (u[1]-mu)*inv*gv.y + bv.y;
    const float o2 = (u[2]-mu)*inv*gv.z + bv.z, o3 = (u[3]-mu)*inv*gv.w + bv.w;
    if (Of) { *(float4*)(Of + (size_t)row * D + tid * 4) = make_float4(o0, o1, o2, o3); }
    if (Oh) {
        v4h hv = { (_Float16)o0, (_Float16)o1, (_Float16)o2, (_Float16)o3 };
        *(v4h*)(Oh + (size_t)row * D + tid * 4) = hv;
    }
}

// ---------------------------------------------------------------------------
// Gate: one block/token, one wave/expert; softmax + top-1 + bucket counts.
// ---------------------------------------------------------------------------
__global__ __launch_bounds__(256) void gate_kernel(
    const _Float16* __restrict__ X, const float* __restrict__ gw,
    const float* __restrict__ gb, const int* __restrict__ idxes,
    int* __restrict__ eid, float* __restrict__ gval, int* __restrict__ counts)
{
    constexpr int D = 1024, E = 8, S = 512;
    const int t = blockIdx.x;
    const int nd = idxes[t / S];
    const int wid = threadIdx.x >> 5, lane = threadIdx.x & 31;
    const _Float16* x = X + (size_t)t * D;
    const float* w = gw + ((size_t)nd * E + wid) * D;
    float s = 0.f;
#pragma unroll
    for (int c = 0; c < 4; ++c) {
        const int base = c * 256 + lane * 8;
        v8h xv = *(const v8h*)(x + base);
        const float4 w0 = *(const float4*)(w + base);
        const float4 w1 = *(const float4*)(w + base + 4);
        s += (float)xv[0]*w0.x + (float)xv[1]*w0.y + (float)xv[2]*w0.z + (float)xv[3]*w0.w
           + (float)xv[4]*w1.x + (float)xv[5]*w1.y + (float)xv[6]*w1.z + (float)xv[7]*w1.w;
    }
    s += sxor<16>(s); s += sxor<8>(s); s += sxor<4>(s); s += sxor<2>(s); s += sxor<1>(s);
    __shared__ float lg[8];
    if (lane == 0) lg[wid] = s + gb[nd * E + wid];
    __syncthreads();
    if (threadIdx.x == 0) {
        float mx = lg[0]; int am = 0;
        for (int e = 1; e < 8; ++e) if (lg[e] > mx) { mx = lg[e]; am = e; }
        float den = 0.f;
        for (int e = 0; e < 8; ++e) den += __expf(lg[e] - mx);
        eid[t] = am;
        gval[t] = 1.f / den;   // max softmax prob
        atomicAdd(&counts[am], 1);
    }
}

__global__ void zero_i32(int* p, int n) {
    if ((int)threadIdx.x < n) p[threadIdx.x] = 0;
}

__global__ void moe_scan_kernel(const int* __restrict__ counts, int* __restrict__ off,
                                int* __restrict__ pc, int* __restrict__ cur,
                                int* __restrict__ perm)
{
    __shared__ int so[9];
    if (threadIdx.x == 0) {
        int o = 0;
        for (int e = 0; e < 8; ++e) {
            so[e] = o;
            o += (counts[e] + 127) & ~127;   // pad buckets to 128 rows
        }
        so[8] = o;
    }
    __syncthreads();
    if (threadIdx.x < 8) {
        const int e = threadIdx.x;
        off[e] = so[e];
        cur[e] = so[e];
        pc[e]  = so[e + 1] - so[e];
    }
    const int tot = so[8];
    for (int i = threadIdx.x; i < tot; i += blockDim.x) perm[i] = -1;
}

__global__ void moe_scatter_kernel(const int* __restrict__ eid, int* __restrict__ cur,
                                   int* __restrict__ perm, int T)
{
    const int t = blockIdx.x * blockDim.x + threadIdx.x;
    if (t < T) {
        const int e = eid[t];
        perm[atomicAdd(&cur[e], 1)] = t;
    }
}

__global__ void cvt_f32_f16(const float* __restrict__ in, _Float16* __restrict__ out, int n4) {
    const int i = blockIdx.x * blockDim.x + threadIdx.x;
    if (i < n4) {
        const float4 v = ((const float4*)in)[i];
        v4h h = { (_Float16)v.x, (_Float16)v.y, (_Float16)v.z, (_Float16)v.w };
        ((v4h*)out)[i] = h;
    }
}

// ---------------------------------------------------------------------------
// Host orchestration
// ---------------------------------------------------------------------------
extern "C" void kernel_launch(void* const* d_in, const int* in_sizes, int n_in,
                              void* d_out, int out_size, void* d_ws, size_t ws_size,
                              hipStream_t stream)
{
    (void)in_sizes; (void)n_in; (void)out_size; (void)ws_size;
    constexpr int cB = 8, cS = 512, cSE = 512, cD = 1024, cH = 16, cF1 = 4096, cI = 1024, cE = 8;
    constexpr int cT = cB * cS;
    constexpr size_t TD  = (size_t)cT * cD;
    constexpr size_t DD  = (size_t)cD * cD;
    constexpr size_t F1D = (size_t)cF1 * cD;
    constexpr size_t EID_W = (size_t)cE * cI * cD;

    const float* hs    = (const float*)d_in[0];
    const float* enc   = (const float*)d_in[1];
    const int*   idxes = (const int*)  d_in[2];
    const float* sa_qw = (const float*)d_in[3];  const float* sa_qb = (const float*)d_in[4];
    const float* sa_kw = (const float*)d_in[5];  const float* sa_kb = (const float*)d_in[6];
    const float* sa_vw = (const float*)d_in[7];  const float* sa_vb = (const float*)d_in[8];
    const float* sa_ow = (const float*)d_in[9];  const float* sa_ob = (const float*)d_in[10];
    const float* ca_qw = (const float*)d_in[11]; const float* ca_qb = (const float*)d_in[12];
    const float* ca_kw = (const float*)d_in[13]; const float* ca_kb = (const float*)d_in[14];
    const float* ca_vw = (const float*)d_in[15]; const float* ca_vb = (const float*)d_in[16];
    const float* ca_ow = (const float*)d_in[17]; const float* ca_ob = (const float*)d_in[18];
    const float* sa_ln_g = (const float*)d_in[19], *sa_ln_b = (const float*)d_in[20];
    const float* ca_ln_g = (const float*)d_in[21], *ca_ln_b = (const float*)d_in[22];
    const float* fin_ln_g = (const float*)d_in[23], *fin_ln_b = (const float*)d_in[24];
    const float* fc1_w = (const float*)d_in[25]; const float* fc1_b = (const float*)d_in[26];
    const float* fc2_w = (const float*)d_in[27]; const float* fc2_b = (const float*)d_in[28];
    const float* exp1_w = (const float*)d_in[29]; const float* exp1_b = (const float*)d_in[30];
    const float* exp2_w = (const float*)d_in[31];
    const float* gate_w = (const float*)d_in[32]; const float* gate_b = (const float*)d_in[33];

    char* ws = (char*)d_ws;
    size_t off = 0;
    auto alloc = [&](size_t bytes) -> void* {
        void* p = ws + off;
        off += (bytes + 255) & ~(size_t)255;
        return p;
    };
    // activations
    _Float16* XB   = (_Float16*)alloc(TD * 2);  // x0 f16 -> h f16 -> x f16 (reused)
    _Float16* ENCB = (_Float16*)alloc(TD * 2);
    _Float16* QB   = (_Float16*)alloc(TD * 2);
    _Float16* KB   = (_Float16*)alloc(TD * 2);
    _Float16* VTB  = (_Float16*)alloc(TD * 2);  // V pre-transposed [B][H][dh][S]
    _Float16* AB   = (_Float16*)alloc(TD * 2);
    float*    PF   = (float*)   alloc(TD * 4);
    float*    HF   = (float*)   alloc(TD * 4);
    float*    XF   = (float*)   alloc(TD * 4);
    _Float16* H1B  = (_Float16*)alloc((size_t)cT * cF1 * 2);
    float*    MO   = (float*)   alloc(TD * 4);
    _Float16* H2B  = (_Float16*)alloc((size_t)cT * cI * 2);
    // f16 weights (pre-converted once per call so GEMM tiles go async->LDS)
    _Float16* WSQ = (_Float16*)alloc(DD * 2);  _Float16* WSK = (_Float16*)alloc(DD * 2);
    _Float16* WSV = (_Float16*)alloc(DD * 2);  _Float16* WSO = (_Float16*)alloc(DD * 2);
    _Float16* WCQ = (_Float16*)alloc(DD * 2);  _Float16* WCK = (_Float16*)alloc(DD * 2);
    _Float16* WCV = (_Float16*)alloc(DD * 2);  _Float16* WCO = (_Float16*)alloc(DD * 2);
    _Float16* WF1 = (_Float16*)alloc(F1D * 2); _Float16* WF2 = (_Float16*)alloc(F1D * 2);
    _Float16* WE1 = (_Float16*)alloc(EID_W * 2); _Float16* WE2 = (_Float16*)alloc(EID_W * 2);
    // MoE bookkeeping
    int* EIDX   = (int*)  alloc((size_t)cT * 4);
    float* GVAL = (float*)alloc((size_t)cT * 4);
    int* COUNTS = (int*)  alloc(64);
    int* OFFS   = (int*)  alloc(64);
    int* PC     = (int*)  alloc(64);
    int* CUR    = (int*)  alloc(64);
    int* PERM   = (int*)  alloc((size_t)(cT + cE * 128) * 4);

    const dim3 blk(256);
    const dim3 gD(cD / 128, cT / 128);
    const dim3 gF(cF1 / 128, cT / 128);
    const float qscale = 0.125f;   // dh^-0.5, dh = 64

    // ---- conversions ----
    const int nTD4 = (int)(TD / 4), nDD4 = (int)(DD / 4);
    const int nF14 = (int)(F1D / 4), nE4 = (int)(EID_W / 4);
    cvt_f32_f16<<<nTD4 / 256, blk, 0, stream>>>(hs,  XB,   nTD4);
    cvt_f32_f16<<<nTD4 / 256, blk, 0, stream>>>(enc, ENCB, nTD4);
    cvt_f32_f16<<<nDD4 / 256, blk, 0, stream>>>(sa_qw, WSQ, nDD4);
    cvt_f32_f16<<<nDD4 / 256, blk, 0, stream>>>(sa_kw, WSK, nDD4);
    cvt_f32_f16<<<nDD4 / 256, blk, 0, stream>>>(sa_vw, WSV, nDD4);
    cvt_f32_f16<<<nDD4 / 256, blk, 0, stream>>>(sa_ow, WSO, nDD4);
    cvt_f32_f16<<<nDD4 / 256, blk, 0, stream>>>(ca_qw, WCQ, nDD4);
    cvt_f32_f16<<<nDD4 / 256, blk, 0, stream>>>(ca_kw, WCK, nDD4);
    cvt_f32_f16<<<nDD4 / 256, blk, 0, stream>>>(ca_vw, WCV, nDD4);
    cvt_f32_f16<<<nDD4 / 256, blk, 0, stream>>>(ca_ow, WCO, nDD4);
    cvt_f32_f16<<<nF14 / 256, blk, 0, stream>>>(fc1_w, WF1, nF14);
    cvt_f32_f16<<<nF14 / 256, blk, 0, stream>>>(fc2_w, WF2, nF14);
    cvt_f32_f16<<<nE4 / 256,  blk, 0, stream>>>(exp1_w, WE1, nE4);
    cvt_f32_f16<<<nE4 / 256,  blk, 0, stream>>>(exp2_w, WE2, nE4);

    // ---- self-attention ----
    gemm_nt_kernel<false,false><<<gD, blk, 0, stream>>>(XB, WSQ, sa_qb, nullptr, QB, cT, cD, cD, qscale);
    gemm_nt_kernel<false,false><<<gD, blk, 0, stream>>>(XB, WSK, sa_kb, nullptr, KB, cT, cD, cD, 1.f);
    gemm_nt_kernel<false,true ><<<gD, blk, 0, stream>>>(XB, WSV, sa_vb, nullptr, VTB, cT, cD, cD, 1.f);
    flash_kernel<<<dim3(cS / 128, cB * cH), blk, 0, stream>>>(QB, KB, VTB, AB, cS, cS);
    gemm_nt_kernel<false,false><<<gD, blk, 0, stream>>>(AB, WSO, sa_ob, PF, nullptr, cT, cD, cD, 1.f);
    ln_kernel<false><<<cT, blk, 0, stream>>>(hs, PF, nullptr, sa_ln_g, sa_ln_b, HF, XB);

    // ---- cross-attention ----
    gemm_nt_kernel<false,false><<<gD, blk, 0, stream>>>(XB,   WCQ, ca_qb, nullptr, QB, cT, cD, cD, qscale);
    gemm_nt_kernel<false,false><<<gD, blk, 0, stream>>>(ENCB, WCK, ca_kb, nullptr, KB, cT, cD, cD, 1.f);
    gemm_nt_kernel<false,true ><<<gD, blk, 0, stream>>>(ENCB, WCV, ca_vb, nullptr, VTB, cT, cD, cD, 1.f);
    flash_kernel<<<dim3(cS / 128, cB * cH), blk, 0, stream>>>(QB, KB, VTB, AB, cS, cSE);
    gemm_nt_kernel<false,false><<<gD, blk, 0, stream>>>(AB, WCO, ca_ob, PF, nullptr, cT, cD, cD, 1.f);
    ln_kernel<false><<<cT, blk, 0, stream>>>(HF, PF, nullptr, ca_ln_g, ca_ln_b, XF, XB);

    // ---- MoE FFN ----
    zero_i32<<<1, 32, 0, stream>>>(COUNTS, cE);
    gate_kernel<<<cT, blk, 0, stream>>>(XB, gate_w, gate_b, idxes, EIDX, GVAL, COUNTS);
    moe_scan_kernel<<<1, blk, 0, stream>>>(COUNTS, OFFS, PC, CUR, PERM);
    moe_scatter_kernel<<<cT / 256, blk, 0, stream>>>(EIDX, CUR, PERM, cT);
    gemm_nt_kernel<true ,false><<<gF, blk, 0, stream>>>(XB,  WF1, fc1_b, nullptr, H1B, cT, cF1, cD, 1.f);
    gemm_nt_kernel<false,false><<<gD, blk, 0, stream>>>(H1B, WF2, fc2_b, MO, nullptr, cT, cD, cF1, 1.f);
    gemm_moe_kernel<true, false><<<dim3(cI / 128, cE * 32), blk, 0, stream>>>(
        XB, WE1, exp1_b, nullptr, H2B, cI, cD, PERM, OFFS, PC);
    gemm_moe_kernel<false, true><<<dim3(cD / 128, cE * 32), blk, 0, stream>>>(
        H2B, WE2, nullptr, MO, nullptr, cD, cI, PERM, OFFS, PC);
    ln_kernel<true><<<cT, blk, 0, stream>>>(XF, MO, GVAL, fin_ln_g, fin_ln_b, (float*)d_out, nullptr);
}
